// ConvNet_79121887527613
// MI455X (gfx1250) — compile-verified, matrix operand order checked by reading, not run on hardware
//
#include <hip/hip_runtime.h>
#include <math.h>

typedef __attribute__((ext_vector_type(16))) _Float16 v16h;
typedef __attribute__((ext_vector_type(8)))  float    v8f;

#define DEVINL __device__ __forceinline__

// ---------------------------------------------------------------------------
// WMMA helpers (CDNA5: V_WMMA_F32_16X16X32_F16)
// ---------------------------------------------------------------------------
DEVINL v8f wmma_f16(v16h a, v16h b, v8f c) {
  return __builtin_amdgcn_wmma_f32_16x16x32_f16(
      /*neg_a=*/false, a, /*neg_b=*/false, b,
      /*c_mod=*/(short)0, c, /*reuse_a=*/false, /*reuse_b=*/false);
}

// Load a 16-half fragment for one lane from a row-major [rows][ld] f16 matrix.
// A layout (ISA 7.12.2, 16-bit A 16x32): lane holds row `row`; 32-bit pair q
// holds K = k0 + hi*8 + 2q + (q>=4 ? 8 : 0). B uses the symmetric layout with
// lane = N column and the same K pattern (weights stored as W[n][k]).
DEVINL v16h ldfrag(const _Float16* __restrict__ base, int ld, int row, int k0, int hi) {
  union { unsigned u[8]; v16h h; } f;
  const _Float16* p = base + (size_t)row * ld + k0 + hi * 8;
#pragma unroll
  for (int q = 0; q < 8; ++q) {
    int k = 2 * q + ((q >= 4) ? 8 : 0);
    f.u[q] = *(const unsigned*)(p + k);
  }
  return f.h;
}

DEVINL v16h ldfrag_lds(const _Float16* base, int ld, int row, int k0, int hi) {
  union { unsigned u[8]; v16h h; } f;
  const _Float16* p = base + row * ld + k0 + hi * 8;
#pragma unroll
  for (int q = 0; q < 8; ++q) {
    int k = 2 * q + ((q >= 4) ? 8 : 0);
    f.u[q] = *(const unsigned*)(p + k);
  }
  return f.h;
}

// ---------------------------------------------------------------------------
// Model constants
// ---------------------------------------------------------------------------
#define NB   32      // batch
#define NT   150     // timesteps
#define HID1 512
#define HID2 1024
#define KPAD 96      // 70 padded to multiple of 32
#define MTOT (NB*NT) // 4800 GEMM rows (m = t*32 + b)
#define XSLICE (2*34*34)   // 2312 floats per (b,t) image
#define CELLS1 (30*9*9)    // 2430 conv1 cells per image
#define BLK1   256
#define NBLK1  ((CELLS1 + BLK1 - 1) / BLK1)   // 10 blocks per image

// LIF constants: v_dec = 0.9 v + 0.1 i ; i_dec = 0.8 i

// ---------------------------------------------------------------------------
// Stage 1: conv1 (2->30, 7x7 VALID) + maxpool3 + LIF-ff, fused t-scan per cell.
// x[b,t] slice staged to LDS each step, shared by 256 cell-threads of block.
// ---------------------------------------------------------------------------
__global__ __launch_bounds__(BLK1) void k_conv1_lif0(const float* __restrict__ x,
                                                     const float* __restrict__ w1,
                                                     const float* __restrict__ b1,
                                                     _Float16* __restrict__ z0h) {
  __shared__ float xs[XSLICE];
  int b    = blockIdx.x / NBLK1;
  int cell = (blockIdx.x % NBLK1) * BLK1 + threadIdx.x;
  bool active = cell < CELLS1;
  int pw = 0, ph = 0, c = 0;
  if (active) {
    pw = cell % 9; int r = cell / 9;
    ph = r % 9;    r /= 9;
    c  = r % 30;
  }
  const float* wc = w1 + c * 2 * 49;
  float bias = active ? b1[c] : 0.f;
  float v = 0.f, cur = 0.f;

  for (int t = 0; t < NT; ++t) {
    __syncthreads();  // previous step's LDS reads complete
    const float* xb = x + (size_t)(b * NT + t) * XSLICE;
    for (int i = threadIdx.x; i < XSLICE; i += BLK1) xs[i] = xb[i];
    __syncthreads();

    if (active) {
      float m = -3.4e38f;
#pragma unroll
      for (int py = 0; py < 3; ++py)
#pragma unroll
        for (int px = 0; px < 3; ++px) {
          int oy = ph * 3 + py, ox = pw * 3 + px;
          float s = bias;
          for (int ci = 0; ci < 2; ++ci) {
            const float* xp = xs + ci * 1156 + oy * 34 + ox;
            const float* wp = wc + ci * 49;
            for (int ky = 0; ky < 7; ++ky)
#pragma unroll
              for (int kx = 0; kx < 7; ++kx)
                s = fmaf(xp[ky * 34 + kx], wp[ky * 7 + kx], s);
          }
          m = fmaxf(m, s);
        }
      float vd = 0.9f * v + 0.1f * cur;
      float id = 0.8f * cur;
      float z  = (vd - 1.0f) > 0.f ? 1.f : 0.f;
      v = (1.f - z) * vd;
      cur = id + m;
      z0h[((size_t)(b * NT + t) * 30 + c) * 81 + ph * 9 + pw] = (_Float16)z;
    }
  }
}

// ---------------------------------------------------------------------------
// Stage 2: conv2 (30->70, 5x5 VALID on 9x9) + maxpool3 -> scalar per (b,t,co)
// ---------------------------------------------------------------------------
__global__ void k_conv2(const _Float16* __restrict__ z0h,
                        const float* __restrict__ w2,
                        const float* __restrict__ b2,
                        float* __restrict__ a2) {
  int tid = blockIdx.x * blockDim.x + threadIdx.x;
  const int N = MTOT * 70;
  if (tid >= N) return;
  int co = tid % 70; int m = tid / 70;
  int b = m % NB;    int t = m / NB;
  const _Float16* zb = z0h + ((size_t)(b * NT + t) * 30) * 81;
  const float* wc = w2 + (size_t)co * 30 * 25;
  float best = -3.4e38f;
#pragma unroll
  for (int py = 0; py < 3; ++py)
#pragma unroll
    for (int px = 0; px < 3; ++px) {
      float s = b2[co];
      for (int ci = 0; ci < 30; ++ci) {
        const _Float16* zp = zb + ci * 81 + py * 9 + px;
        const float* wp = wc + ci * 25;
#pragma unroll
        for (int ky = 0; ky < 5; ++ky)
#pragma unroll
          for (int kx = 0; kx < 5; ++kx)
            s = fmaf((float)zp[ky * 9 + kx], wp[ky * 5 + kx], s);
      }
      best = fmaxf(best, s);
    }
  a2[(size_t)m * 70 + co] = best;
}

// ---------------------------------------------------------------------------
// Stage 3: LIF-ff scan over t for the 70-d feature vector -> f16 spikes (K-pad)
// ---------------------------------------------------------------------------
__global__ void k_lif1(const float* __restrict__ a2, _Float16* __restrict__ z2h) {
  int tid = blockIdx.x * blockDim.x + threadIdx.x;
  if (tid >= NB * 70) return;
  int co = tid % 70; int b = tid / 70;
  float v = 0.f, cur = 0.f;
  for (int t = 0; t < NT; ++t) {
    float inp = a2[(size_t)(t * NB + b) * 70 + co];
    float vd = 0.9f * v + 0.1f * cur;
    float id = 0.8f * cur;
    float z  = (vd - 1.0f) > 0.f ? 1.f : 0.f;
    v = (1.f - z) * vd;
    cur = id + inp;
    z2h[(size_t)(t * NB + b) * KPAD + co] = (_Float16)z;
  }
}

// ---------------------------------------------------------------------------
// Utility kernels: zero f16, f32->f16 convert, f32->f16 convert with K padding
// ---------------------------------------------------------------------------
__global__ void k_zero_h(_Float16* p, int n) {
  int tid = blockIdx.x * blockDim.x + threadIdx.x;
  if (tid < n) p[tid] = (_Float16)0.f;
}
__global__ void k_cvt(_Float16* __restrict__ dst, const float* __restrict__ src, int n) {
  int tid = blockIdx.x * blockDim.x + threadIdx.x;
  if (tid < n) dst[tid] = (_Float16)src[tid];
}
__global__ void k_cvt_pad(_Float16* __restrict__ dst, const float* __restrict__ src, int rows) {
  int tid = blockIdx.x * blockDim.x + threadIdx.x;
  if (tid >= rows * KPAD) return;
  int c = tid % KPAD; int r = tid / KPAD;
  dst[tid] = (c < 70) ? (_Float16)src[r * 70 + c] : (_Float16)0.f;
}

// ---------------------------------------------------------------------------
// Time-parallel input-projection GEMM: C[m,n] = sum_k A[m,k] * W[n,k]
// A: [MTOT][KPAD] f16, W: [N][KPAD] f16, C: [MTOT][N] f32. One wave per tile.
// ---------------------------------------------------------------------------
__global__ void k_gemm_inproj(float* __restrict__ C,
                              const _Float16* __restrict__ A,
                              const _Float16* __restrict__ W,
                              int ntN, int mTiles, int ldc) {
  int wid  = (blockIdx.x * blockDim.x + threadIdx.x) >> 5;
  int lane = threadIdx.x & 31;
  int lo = lane & 15, hi = lane >> 4;
  int nt = wid % ntN, mt = wid / ntN;
  if (mt >= mTiles) return;
  v8f acc = {0.f, 0.f, 0.f, 0.f, 0.f, 0.f, 0.f, 0.f};
#pragma unroll
  for (int k0 = 0; k0 < KPAD; k0 += 32) {
    v16h a = ldfrag(A, KPAD, mt * 16 + lo, k0, hi);
    v16h b = ldfrag(W, KPAD, nt * 16 + lo, k0, hi);
    acc = wmma_f16(a, b, acc);
  }
#pragma unroll
  for (int r = 0; r < 8; ++r)
    C[(size_t)(mt * 16 + hi * 8 + r) * ldc + nt * 16 + lo] = acc[r];
}

// ---------------------------------------------------------------------------
// Persistent recurrent LIF layer body: one workgroup (32 waves), z_prev in LDS.
// Per step: rec = z_prev @ Wrec.T via WMMA, cur = rec + P[t], LIF epilogue on
// the accumulator layout, new spikes written to LDS + global (f16).
// ---------------------------------------------------------------------------
template <int HID, int NTPW>
DEVINL void rec_body(const _Float16* __restrict__ Wr,
                     const float* __restrict__ P,
                     _Float16* __restrict__ Zout,
                     _Float16* zprev /* LDS, >= 32*HID halves */) {
  for (int i = threadIdx.x; i < 32 * HID; i += 1024) zprev[i] = (_Float16)0.f;
  __syncthreads();

  int w = threadIdx.x >> 5, lane = threadIdx.x & 31;
  int lo = lane & 15, hi = lane >> 4;
  int mt = w & 1;        // 2 M-tiles (B=32)
  int nc = w >> 1;       // 16 N-column groups, NTPW tiles each (stride 16 tiles)
  int rowA = mt * 16 + lo;

  float vst[NTPW][8] = {};
  float ist[NTPW][8] = {};
  const v8f z8 = {0.f, 0.f, 0.f, 0.f, 0.f, 0.f, 0.f, 0.f};

  for (int t = 0; t < NT; ++t) {
    v8f acc[NTPW];
#pragma unroll
    for (int j = 0; j < NTPW; ++j) acc[j] = z8;

    for (int k0 = 0; k0 < HID; k0 += 32) {
      v16h a = ldfrag_lds(zprev, HID, rowA, k0, hi);
#pragma unroll
      for (int j = 0; j < NTPW; ++j) {
        v16h b = ldfrag(Wr, HID, (nc + 16 * j) * 16 + lo, k0, hi);
        acc[j] = wmma_f16(a, b, acc[j]);
      }
    }
    __syncthreads();  // all waves finished reading z_{t-1} from LDS

#pragma unroll
    for (int j = 0; j < NTPW; ++j) {
      int n_ = (nc + 16 * j) * 16 + lo;
#pragma unroll
      for (int r = 0; r < 8; ++r) {
        int b_ = mt * 16 + hi * 8 + r;
        float cur = acc[j][r] + P[(size_t)(t * NB + b_) * HID + n_];
        float vd = 0.9f * vst[j][r] + 0.1f * ist[j][r];
        float id = 0.8f * ist[j][r];
        float z  = (vd - 0.4f) > 0.f ? 1.f : 0.f;
        vst[j][r] = (1.f - z) * vd;
        ist[j][r] = id + cur;
        _Float16 zh = (_Float16)z;
        zprev[b_ * HID + n_] = zh;
        Zout[(size_t)(t * NB + b_) * HID + n_] = zh;
      }
    }
    __syncthreads();  // z_t visible before next step's GEMM
  }
}

// Both recurrent layers in one launch: block 0 -> l1 (512), block 1 -> cl (1024).
// Block-uniform branch, so the two persistent workgroups run on separate WGPs
// concurrently instead of serializing on the stream.
__global__ __launch_bounds__(1024) void k_rec_both(const _Float16* __restrict__ w1h,
                                                   const float* __restrict__ P1,
                                                   _Float16* __restrict__ Z1,
                                                   const _Float16* __restrict__ w2h,
                                                   const float* __restrict__ Pc,
                                                   _Float16* __restrict__ Zc) {
  __shared__ _Float16 zprev[32 * HID2];  // 64 KB, sized for the larger layer
  if (blockIdx.x == 0)
    rec_body<HID1, 2>(w1h, P1, Z1, zprev);
  else
    rec_body<HID2, 4>(w2h, Pc, Zc, zprev);
}

// ---------------------------------------------------------------------------
// Readout projection (time-parallel), LI scan + max-over-t, log-softmax
// ---------------------------------------------------------------------------
__global__ void k_proj(float* __restrict__ out, const _Float16* __restrict__ Z,
                       const float* __restrict__ W, int O, int hid) {
  int tid = blockIdx.x * blockDim.x + threadIdx.x;
  if (tid >= MTOT * O) return;
  int o = tid % O; int m = tid / O;
  const _Float16* zp = Z + (size_t)m * hid;
  const float* wp = W + (size_t)o * hid;
  float s = 0.f;
  for (int n = 0; n < hid; ++n) s = fmaf((float)zp[n], wp[n], s);
  out[tid] = s;
}

__global__ void k_li_scan(const float* __restrict__ proj, int O, float* __restrict__ outp) {
  int tid = blockIdx.x * blockDim.x + threadIdx.x;
  if (tid >= NB * O) return;
  int o = tid % O; int b = tid / O;
  float v = 0.f, cur = 0.f, best = -3.4e38f;
  for (int t = 0; t < NT; ++t) {
    float inp = proj[(size_t)(t * NB + b) * O + o];
    float vn = 0.9f * v + 0.1f * cur;
    cur = 0.8f * cur + inp;
    v = vn;
    best = fmaxf(best, vn);
  }
  outp[b * O + o] = best;
}

__global__ void k_logsoftmax(const float* __restrict__ vmaxc, float* __restrict__ out) {
  int b = blockIdx.x * blockDim.x + threadIdx.x;
  if (b >= NB) return;
  float x[10];
  float m = -3.4e38f;
  for (int j = 0; j < 10; ++j) { x[j] = vmaxc[b * 10 + j]; m = fmaxf(m, x[j]); }
  float s = 0.f;
  for (int j = 0; j < 10; ++j) s += expf(x[j] - m);
  float l = logf(s);
  for (int j = 0; j < 10; ++j) out[NB * 4 + b * 10 + j] = x[j] - m - l;
}

// ---------------------------------------------------------------------------
// Host-side launcher
// ---------------------------------------------------------------------------
static inline size_t alignup(size_t v) { return (v + 255) & ~(size_t)255; }

extern "C" void kernel_launch(void* const* d_in, const int* in_sizes, int n_in,
                              void* d_out, int out_size, void* d_ws, size_t ws_size,
                              hipStream_t stream) {
  (void)in_sizes; (void)n_in; (void)out_size; (void)ws_size;
  const float* x        = (const float*)d_in[0];
  const float* conv1_w  = (const float*)d_in[1];
  const float* conv1_b  = (const float*)d_in[2];
  const float* conv2_w  = (const float*)d_in[3];
  const float* conv2_b  = (const float*)d_in[4];
  const float* l1_in_w  = (const float*)d_in[5];
  const float* l1_rec_w = (const float*)d_in[6];
  const float* fc_out_w = (const float*)d_in[7];
  const float* cl_in_w  = (const float*)d_in[8];
  const float* cl_rec_w = (const float*)d_in[9];
  const float* cl_fc_w  = (const float*)d_in[10];
  float* out = (float*)d_out;

  // workspace carve-up
  char* p = (char*)d_ws;
  size_t off = 0;
  auto take = [&](size_t bytes) { char* q = p + off; off = alignup(off + bytes); return q; };
  _Float16* z0h    = (_Float16*)take((size_t)NB * NT * 30 * 81 * 2);  // 23.3 MB
  float*    a2     = (float*)   take((size_t)MTOT * 70 * 4);
  _Float16* z2h    = (_Float16*)take((size_t)MTOT * KPAD * 2);
  float*    P1     = (float*)   take((size_t)MTOT * HID1 * 4);
  float*    Pc     = (float*)   take((size_t)MTOT * HID2 * 4);
  _Float16* Z1h    = (_Float16*)take((size_t)MTOT * HID1 * 2);
  _Float16* Zch    = (_Float16*)take((size_t)MTOT * HID2 * 2);
  _Float16* wrec_h = (_Float16*)take((size_t)HID1 * HID1 * 2);
  _Float16* crec_h = (_Float16*)take((size_t)HID2 * HID2 * 2);
  _Float16* l1in_h = (_Float16*)take((size_t)HID1 * KPAD * 2);
  _Float16* clin_h = (_Float16*)take((size_t)HID2 * KPAD * 2);
  float*    proj_o = (float*)   take((size_t)MTOT * 4 * 4);
  float*    proj_c = (float*)   take((size_t)MTOT * 10 * 4);
  float*    vmaxc  = (float*)   take((size_t)NB * 10 * 4);

  const int TB = 256;
  auto blks = [](int n, int tb) { return (n + tb - 1) / tb; };

  // weight conversions (f32 -> f16, K-padded where needed)
  k_cvt    <<<blks(HID1 * HID1, TB), TB, 0, stream>>>(wrec_h, l1_rec_w, HID1 * HID1);
  k_cvt    <<<blks(HID2 * HID2, TB), TB, 0, stream>>>(crec_h, cl_rec_w, HID2 * HID2);
  k_cvt_pad<<<blks(HID1 * KPAD, TB), TB, 0, stream>>>(l1in_h, l1_in_w, HID1);
  k_cvt_pad<<<blks(HID2 * KPAD, TB), TB, 0, stream>>>(clin_h, cl_in_w, HID2);
  k_zero_h <<<blks(MTOT * KPAD, TB), TB, 0, stream>>>(z2h, MTOT * KPAD);

  // front-end: conv1+LIF0 (fused t-scan, LDS-staged x slices), conv2, LIF1
  k_conv1_lif0<<<NB * NBLK1, BLK1, 0, stream>>>(x, conv1_w, conv1_b, z0h);
  k_conv2     <<<blks(MTOT * 70, TB), TB, 0, stream>>>(z0h, conv2_w, conv2_b, a2);
  k_lif1      <<<blks(NB * 70,   TB), TB, 0, stream>>>(a2, z2h);

  // time-parallel input projections (WMMA)
  {
    int ntN = HID1 / 16, waves = (MTOT / 16) * ntN;
    k_gemm_inproj<<<blks(waves * 32, TB), TB, 0, stream>>>(P1, z2h, l1in_h, ntN, MTOT / 16, HID1);
  }
  {
    int ntN = HID2 / 16, waves = (MTOT / 16) * ntN;
    k_gemm_inproj<<<blks(waves * 32, TB), TB, 0, stream>>>(Pc, z2h, clin_h, ntN, MTOT / 16, HID2);
  }

  // both persistent recurrent layers concurrently (2 workgroups, 2 WGPs)
  k_rec_both<<<2, 1024, 0, stream>>>(wrec_h, P1, Z1h, crec_h, Pc, Zch);

  // readouts
  k_proj   <<<blks(MTOT * 4,  TB), TB, 0, stream>>>(proj_o, Z1h, fc_out_w, 4,  HID1);
  k_proj   <<<blks(MTOT * 10, TB), TB, 0, stream>>>(proj_c, Zch, cl_fc_w, 10, HID2);
  k_li_scan<<<blks(NB * 4,  128), 128, 0, stream>>>(proj_o, 4,  out);    // y_hat -> out[0:128]
  k_li_scan<<<blks(NB * 10, 128), 128, 0, stream>>>(proj_c, 10, vmaxc);
  k_logsoftmax<<<1, 32, 0, stream>>>(vmaxc, out);                        // out[128:448]
}
